// SpikeTransformerHierarchical_2602750181577
// MI455X (gfx1250) — compile-verified, matrix operand order checked by reading, not run on hardware
//
#include <hip/hip_runtime.h>
#include <math.h>

typedef __attribute__((ext_vector_type(16))) _Float16 v16h;
typedef __attribute__((ext_vector_type(8)))  float    v8f;

#define S_LEN 256
#define E_DIM 64

// ---------------------------------------------------------------------------
// Encoder: one wave (32 lanes = 32 time steps) per sample n = b*S + s.
// conv1 (6->32, k=5, pad2) + relu ; conv2 (32->64, k=3, pad1) + relu ; mean(T)
// then + shank_emb[g] + pos_encoding(s), zeroed if masked.
// ---------------------------------------------------------------------------
#define ENC_WPB 4
__global__ void encoder_kernel(const float* __restrict__ wf,
                               const int* __restrict__ sid,
                               const unsigned char* __restrict__ maskb,
                               const float* __restrict__ w1, const float* __restrict__ b1,
                               const float* __restrict__ w2, const float* __restrict__ b2,
                               const float* __restrict__ semb,
                               float* __restrict__ h)
{
    __shared__ float xs[ENC_WPB][6][36];     // padded time: idx t+2, pads at 0,1,34,35
    __shared__ float h1s[ENC_WPB][32][34];   // padded time: idx t+1, pads at 0,33

    const int wv   = threadIdx.x >> 5;
    const int lane = threadIdx.x & 31;
    const int n    = blockIdx.x * ENC_WPB + wv;      // sample index < 32768
    const int s    = n & (S_LEN - 1);
    const int g    = sid[n];

    // stage input window
    for (int c = 0; c < 6; ++c)
        xs[wv][c][lane + 2] = wf[((size_t)n * 6 + c) * 32 + lane];
    if (lane < 4) {
        int p = (lane < 2) ? lane : 32 + lane;       // 0,1,34,35
        for (int c = 0; c < 6; ++c) xs[wv][c][p] = 0.f;
    }
    __syncthreads();

    // conv1 + relu
    for (int o = 0; o < 32; ++o) {
        float acc = b1[g * 32 + o];
        #pragma unroll
        for (int c = 0; c < 6; ++c)
            #pragma unroll
            for (int k = 0; k < 5; ++k)
                acc += w1[((g * 32 + o) * 6 + c) * 5 + k] * xs[wv][c][lane + k];
        h1s[wv][o][lane + 1] = fmaxf(acc, 0.f);
    }
    if (lane < 2) for (int o = 0; o < 32; ++o) h1s[wv][o][lane ? 33 : 0] = 0.f;
    __syncthreads();

    // conv2 + relu + mean over T (wave reduction)
    float r0 = 0.f, r1 = 0.f;
    for (int e = 0; e < 64; ++e) {
        float acc = b2[g * 64 + e];
        for (int o = 0; o < 32; ++o) {
            const float* w = &w2[((g * 64 + e) * 32 + o) * 3];
            acc += w[0] * h1s[wv][o][lane]
                 + w[1] * h1s[wv][o][lane + 1]
                 + w[2] * h1s[wv][o][lane + 2];
        }
        acc = fmaxf(acc, 0.f);
        #pragma unroll
        for (int off = 16; off > 0; off >>= 1) acc += __shfl_xor(acc, off, 32);
        if ((e & 31) == lane) { if (e < 32) r0 = acc; else r1 = acc; }
    }

    const bool active = (maskb[n] == 0);
    #pragma unroll
    for (int hf = 0; hf < 2; ++hf) {
        int e = lane + hf * 32;
        float v = active ? ((hf ? r1 : r0) * (1.f / 32.f)) : 0.f;
        float div = __expf(-(float)(e & ~1) * (9.210340372f / 64.f)); // ln(1e4)/dim
        float ang = (float)s * div;
        v += semb[g * 64 + e] + ((e & 1) ? __cosf(ang) : __sinf(ang));
        h[(size_t)n * 64 + e] = v;
    }
}

// ---------------------------------------------------------------------------
// Weight prep: f32 W[N][K]  ->  f16 B[K][N]  (B-matrix for C = A * W^T)
// ---------------------------------------------------------------------------
__global__ void convert_w(const float* __restrict__ W, _Float16* __restrict__ out,
                          int N, int K)
{
    int idx = blockIdx.x * blockDim.x + threadIdx.x;
    if (idx >= N * K) return;
    int nn = idx / K, kk = idx - nn * K;
    out[(size_t)kk * N + nn] = (_Float16)W[(size_t)nn * K + kk];
}

// ---------------------------------------------------------------------------
// Generic WMMA GEMM: C[M,N] = A(f32->f16)[M,K] * B(f16)[K,N] + bias, opt relu,
// output f32 or f16 (compile-time). One wave per 16x16 C tile; K multiple of 32.
// A-fragment k-mapping (16-bit A 16x32): lane<16 -> kbase 0, lane>=16 -> kbase 8;
// halves 0..7 -> K = kbase..kbase+7 ; halves 8..15 -> K = 16+kbase..16+kbase+7.
// ---------------------------------------------------------------------------
template <int RELU, int OUTF16>
__global__ void gemm_wmma(const float* __restrict__ A, const _Float16* __restrict__ B,
                          const float* __restrict__ bias, void* __restrict__ Cout,
                          int M, int N, int K)
{
    const int wave  = blockIdx.x * (blockDim.x >> 5) + (threadIdx.x >> 5);
    const int lane  = threadIdx.x & 31;
    const int ntile = N >> 4;
    const int tm = wave / ntile, tn = wave - tm * ntile;
    if (tm >= (M >> 4)) return;                       // wave-uniform

    const int m     = tm * 16 + (lane & 15);
    const int n     = tn * 16 + (lane & 15);
    const int kbase = (lane & 16) ? 8 : 0;

    v8f c = {};
    for (int k0 = 0; k0 < K; k0 += 32) {
        const float* arow = A + (size_t)m * K + k0;
        // two contiguous 8-float groups -> vector loads
        float4 alo0 = *(const float4*)(arow + kbase);
        float4 alo1 = *(const float4*)(arow + kbase + 4);
        float4 ahi0 = *(const float4*)(arow + 16 + kbase);
        float4 ahi1 = *(const float4*)(arow + 16 + kbase + 4);
        v16h a, b;
        a[0]  = (_Float16)alo0.x; a[1]  = (_Float16)alo0.y;
        a[2]  = (_Float16)alo0.z; a[3]  = (_Float16)alo0.w;
        a[4]  = (_Float16)alo1.x; a[5]  = (_Float16)alo1.y;
        a[6]  = (_Float16)alo1.z; a[7]  = (_Float16)alo1.w;
        a[8]  = (_Float16)ahi0.x; a[9]  = (_Float16)ahi0.y;
        a[10] = (_Float16)ahi0.z; a[11] = (_Float16)ahi0.w;
        a[12] = (_Float16)ahi1.x; a[13] = (_Float16)ahi1.y;
        a[14] = (_Float16)ahi1.z; a[15] = (_Float16)ahi1.w;
        #pragma unroll
        for (int i = 0; i < 16; ++i) {
            int kk = (i < 8) ? (kbase + i) : (16 + kbase + (i - 8));
            b[i] = B[(size_t)(k0 + kk) * N + n];
        }
        c = __builtin_amdgcn_wmma_f32_16x16x32_f16(false, a, false, b,
                                                   (short)0, c, false, false);
    }
    const float bs = bias[n];
    #pragma unroll
    for (int j = 0; j < 8; ++j) {
        int mm = tm * 16 + j + ((lane & 16) ? 8 : 0);
        float v = c[j] + bs;
        if (RELU) v = fmaxf(v, 0.f);
        if (OUTF16) ((_Float16*)Cout)[(size_t)mm * N + n] = (_Float16)v;
        else        ((float*)Cout)[(size_t)mm * N + n]    = v;
    }
}

// ---------------------------------------------------------------------------
// Attention: one wave per (b, head, 16-query tile). qkv f16 [B*S, 192]:
// q cols 0..63, k 64..127, v 128..191 ; head h slice = h*16..h*16+15.
// dh=16 zero-padded to K=32 for WMMA. Scores f32 softmax in LDS.
// ---------------------------------------------------------------------------
__global__ void attn_kernel(const _Float16* __restrict__ qkv,
                            const unsigned char* __restrict__ maskb,
                            float* __restrict__ Obuf)
{
    __shared__ float sc[16][264];

    const int blk  = blockIdx.x;
    const int qt   = blk & 15;
    const int hh   = (blk >> 4) & 3;
    const int b    = blk >> 6;
    const int lane = threadIdx.x;
    const int kbase = (lane & 16) ? 8 : 0;
    const int nloc  = lane & 15;              // B-matrix column / C column
    const int mrow  = lane & 15;              // A-matrix row
    const int mC    = ((lane & 16) ? 8 : 0);  // C-row offset

    // Q fragment (16 x 32, upper 16 K zero-padded)
    v16h aq;
    #pragma unroll
    for (int i = 0; i < 16; ++i) {
        int kk = (i < 8) ? (kbase + i) : (16 + kbase + (i - 8));
        aq[i] = (kk < 16)
            ? qkv[((size_t)(b * S_LEN + qt * 16 + mrow)) * 192 + hh * 16 + kk]
            : (_Float16)0.f;
    }

    // scores = Q K^T * 0.25 + key bias
    for (int kt = 0; kt < 16; ++kt) {
        v16h bk;
        #pragma unroll
        for (int i = 0; i < 16; ++i) {
            int kk = (i < 8) ? (kbase + i) : (16 + kbase + (i - 8));
            bk[i] = (kk < 16)
                ? qkv[((size_t)(b * S_LEN + kt * 16 + nloc)) * 192 + 64 + hh * 16 + kk]
                : (_Float16)0.f;
        }
        v8f sfrag = {};
        sfrag = __builtin_amdgcn_wmma_f32_16x16x32_f16(false, aq, false, bk,
                                                       (short)0, sfrag, false, false);
        const int key = kt * 16 + nloc;
        const float biasv = maskb[b * S_LEN + key] ? -1e9f : 0.f;
        #pragma unroll
        for (int j = 0; j < 8; ++j)
            sc[j + mC][key] = sfrag[j] * 0.25f + biasv;   // 1/sqrt(dh=16)
    }
    __syncthreads();

    // row-wise softmax over 256 keys
    for (int r = 0; r < 16; ++r) {
        float mx = -1e30f;
        for (int cix = lane; cix < 256; cix += 32) mx = fmaxf(mx, sc[r][cix]);
        #pragma unroll
        for (int o = 16; o > 0; o >>= 1) mx = fmaxf(mx, __shfl_xor(mx, o, 32));
        float sm = 0.f;
        for (int cix = lane; cix < 256; cix += 32) {
            float ex = __expf(sc[r][cix] - mx);
            sc[r][cix] = ex; sm += ex;
        }
        #pragma unroll
        for (int o = 16; o > 0; o >>= 1) sm += __shfl_xor(sm, o, 32);
        float inv = 1.f / sm;
        for (int cix = lane; cix < 256; cix += 32) sc[r][cix] *= inv;
    }
    __syncthreads();

    // O = P V  (8 WMMA steps over 256 keys)
    v8f o8 = {};
    for (int kc = 0; kc < 8; ++kc) {
        v16h ap, bv;
        #pragma unroll
        for (int i = 0; i < 16; ++i) {
            int kk  = (i < 8) ? (kbase + i) : (16 + kbase + (i - 8));
            int key = kc * 32 + kk;
            ap[i] = (_Float16)sc[mrow][key];
            bv[i] = qkv[((size_t)(b * S_LEN + key)) * 192 + 128 + hh * 16 + nloc];
        }
        o8 = __builtin_amdgcn_wmma_f32_16x16x32_f16(false, ap, false, bv,
                                                    (short)0, o8, false, false);
    }
    #pragma unroll
    for (int j = 0; j < 8; ++j)
        Obuf[((size_t)(b * S_LEN + qt * 16 + j + mC)) * 64 + hh * 16 + nloc] = o8[j];
}

// ---------------------------------------------------------------------------
// Residual + LayerNorm over E=64: one wave per row, 2 elems per lane.
// h = LN(h + add) * g + b
// ---------------------------------------------------------------------------
__global__ void ln_kernel(float* __restrict__ h, const float* __restrict__ add,
                          const float* __restrict__ g, const float* __restrict__ be)
{
    const int row  = blockIdx.x * 4 + (threadIdx.x >> 5);
    const int lane = threadIdx.x & 31;
    const size_t base = (size_t)row * 64;
    float x0 = h[base + lane]      + add[base + lane];
    float x1 = h[base + lane + 32] + add[base + lane + 32];
    float sm = x0 + x1;
    #pragma unroll
    for (int o = 16; o > 0; o >>= 1) sm += __shfl_xor(sm, o, 32);
    float mean = sm * (1.f / 64.f);
    float d0 = x0 - mean, d1 = x1 - mean;
    float vs = d0 * d0 + d1 * d1;
    #pragma unroll
    for (int o = 16; o > 0; o >>= 1) vs += __shfl_xor(vs, o, 32);
    float inv = rsqrtf(vs * (1.f / 64.f) + 1e-5f);
    h[base + lane]      = d0 * inv * g[lane]      + be[lane];
    h[base + lane + 32] = d1 * inv * g[lane + 32] + be[lane + 32];
}

// ---------------------------------------------------------------------------
// Masked mean pool: one block (64 threads) per batch element.
// ---------------------------------------------------------------------------
__global__ void pool_kernel(const float* __restrict__ h,
                            const unsigned char* __restrict__ maskb,
                            float* __restrict__ pooled)
{
    const int b = blockIdx.x, e = threadIdx.x;
    float sm = 0.f, cnt = 0.f;
    for (int s = 0; s < S_LEN; ++s) {
        if (!maskb[b * S_LEN + s]) {
            sm += h[((size_t)b * S_LEN + s) * 64 + e];
            cnt += 1.f;
        }
    }
    pooled[b * 64 + e] = sm / (cnt + 1e-8f);
}

// ---------------------------------------------------------------------------
// Output heads: one block (64 threads) per batch element. out row = 16 floats:
// [0..2]=cls, [3..8]=mus, [9..14]=sigmas, [15]=d
// ---------------------------------------------------------------------------
__global__ void heads_kernel(const float* __restrict__ pooled,
    const float* cw1, const float* cb1, const float* cw2, const float* cb2,
    const float* mw1, const float* mb1, const float* mw2, const float* mb2,
    const float* sw1, const float* sb1, const float* sw2, const float* sb2,
    const float* dw1, const float* db1, const float* dw2, const float* db2,
    float* __restrict__ out)
{
    __shared__ float pl[64];
    __shared__ float hid[64];
    const int b = blockIdx.x, j = threadIdx.x;
    pl[j] = pooled[b * 64 + j];
    __syncthreads();

    float a = cb1[j];
    for (int e = 0; e < 64; ++e) a += cw1[j * 64 + e] * pl[e];
    hid[j] = fmaxf(a, 0.f);
    __syncthreads();
    if (j < 3) {
        float v = cb2[j];
        for (int k = 0; k < 64; ++k) v += cw2[j * 64 + k] * hid[k];
        out[b * 16 + j] = v;
    }
    __syncthreads();

    for (int z = 0; z < 3; ++z) {
        float a2 = mb1[z * 64 + j];
        for (int e = 0; e < 64; ++e) a2 += mw1[(z * 64 + j) * 64 + e] * pl[e];
        hid[j] = fmaxf(a2, 0.f);
        __syncthreads();
        if (j < 2) {
            float v = mb2[z * 2 + j];
            for (int k = 0; k < 64; ++k) v += mw2[(z * 2 + j) * 64 + k] * hid[k];
            out[b * 16 + 3 + z * 2 + j] = v;
        }
        __syncthreads();
    }
    for (int z = 0; z < 3; ++z) {
        float a2 = sb1[z * 64 + j];
        for (int e = 0; e < 64; ++e) a2 += sw1[(z * 64 + j) * 64 + e] * pl[e];
        hid[j] = fmaxf(a2, 0.f);
        __syncthreads();
        if (j < 2) {
            float v = sb2[z * 2 + j];
            for (int k = 0; k < 64; ++k) v += sw2[(z * 2 + j) * 64 + k] * hid[k];
            out[b * 16 + 9 + z * 2 + j] = __expf(v);
        }
        __syncthreads();
    }
    float a3 = db1[j];
    for (int e = 0; e < 64; ++e) a3 += dw1[j * 64 + e] * pl[e];
    hid[j] = fmaxf(a3, 0.f);
    __syncthreads();
    if (j == 0) {
        float v = db2[0];
        for (int k = 0; k < 64; ++k) v += dw2[k] * hid[k];
        out[b * 16 + 15] = 1.f / (1.f + __expf(-v));
    }
}

// ---------------------------------------------------------------------------
extern "C" void kernel_launch(void* const* d_in, const int* in_sizes, int n_in,
                              void* d_out, int out_size, void* d_ws, size_t ws_size,
                              hipStream_t stream)
{
    const float*         wfm  = (const float*)d_in[0];
    const int*           sid  = (const int*)d_in[1];
    const unsigned char* msk  = (const unsigned char*)d_in[2];   // jnp bool = 1 byte
    const float* c1w = (const float*)d_in[3];
    const float* c1b = (const float*)d_in[4];
    const float* c2w = (const float*)d_in[5];
    const float* c2b = (const float*)d_in[6];
    const float* semb = (const float*)d_in[7];
    const float* inpw = (const float*)d_in[8];
    const float* inpb = (const float*)d_in[9];
    const float* outw = (const float*)d_in[10];
    const float* outb = (const float*)d_in[11];
    const float* ln1g = (const float*)d_in[12];
    const float* ln1b = (const float*)d_in[13];
    const float* ff1w = (const float*)d_in[14];
    const float* ff1b = (const float*)d_in[15];
    const float* ff2w = (const float*)d_in[16];
    const float* ff2b = (const float*)d_in[17];
    const float* ln2g = (const float*)d_in[18];
    const float* ln2b = (const float*)d_in[19];
    float* out = (float*)d_out;

    char* ws = (char*)d_ws;
    float*     h      = (float*)(ws);                              //  8 MiB
    float*     buf1   = (float*)(ws + (size_t)(8)  * (1u << 20));  // 32 MiB (ff1 / outproj)
    float*     buf2   = (float*)(ws + (size_t)(40) * (1u << 20));  //  8 MiB (attn O / ff2)
    _Float16*  qkv16  = (_Float16*)(ws + (size_t)(48) * (1u << 20)); // 12 MiB
    _Float16*  wf16   = (_Float16*)(ws + (size_t)(60) * (1u << 20)); // 192 KiB
    float*     pooled = (float*)(ws + (size_t)(61) * (1u << 20));    // 32 KiB
    _Float16* w_qkv = wf16;            // 2 * 64*192
    _Float16* w_out = wf16 + 24576;    // 2 * 64*64
    _Float16* w_ff1 = wf16 + 32768;    // 2 * 64*256
    _Float16* w_ff2 = wf16 + 65536;    // 2 * 256*64

    const int M = 128 * S_LEN;         // 32768 rows

    // 1) per-shank conv encoder + embeddings
    encoder_kernel<<<M / ENC_WPB, 32 * ENC_WPB, 0, stream>>>(
        wfm, sid, msk, c1w, c1b, c2w, c2b, semb, h);

    // 2) weight prep (f32 -> f16 B-layout)
    for (int l = 0; l < 2; ++l) {
        convert_w<<<(192 * 64 + 255) / 256, 256, 0, stream>>>(inpw + l * 192 * 64, w_qkv + l * 12288, 192, 64);
        convert_w<<<(64 * 64  + 255) / 256, 256, 0, stream>>>(outw + l * 4096,     w_out + l * 4096,  64,  64);
        convert_w<<<(256 * 64 + 255) / 256, 256, 0, stream>>>(ff1w + l * 16384,    w_ff1 + l * 16384, 256, 64);
        convert_w<<<(64 * 256 + 255) / 256, 256, 0, stream>>>(ff2w + l * 16384,    w_ff2 + l * 16384, 64,  256);
    }

    // 3) transformer layers
    for (int l = 0; l < 2; ++l) {
        int t_qkv = (M / 16) * 12, t_e = (M / 16) * 4, t_ff = (M / 16) * 16;
        gemm_wmma<0, 1><<<(t_qkv + 3) / 4, 128, 0, stream>>>(h, w_qkv + l * 12288, inpb + l * 192,
                                                             qkv16, M, 192, 64);
        attn_kernel<<<128 * 4 * 16, 32, 0, stream>>>(qkv16, msk, buf2);
        gemm_wmma<0, 0><<<(t_e + 3) / 4, 128, 0, stream>>>(buf2, w_out + l * 4096, outb + l * 64,
                                                           buf1, M, 64, 64);
        ln_kernel<<<M / 4, 128, 0, stream>>>(h, buf1, ln1g + l * 64, ln1b + l * 64);
        gemm_wmma<1, 0><<<(t_ff + 3) / 4, 128, 0, stream>>>(h, w_ff1 + l * 16384, ff1b + l * 256,
                                                            buf1, M, 256, 64);
        gemm_wmma<0, 0><<<(t_e + 3) / 4, 128, 0, stream>>>(buf1, w_ff2 + l * 16384, ff2b + l * 64,
                                                           buf2, M, 64, 256);
        ln_kernel<<<M / 4, 128, 0, stream>>>(h, buf2, ln2g + l * 64, ln2b + l * 64);
    }

    // 4) masked mean pool + heads
    pool_kernel<<<128, 64, 0, stream>>>(h, msk, pooled);
    heads_kernel<<<128, 64, 0, stream>>>(pooled,
        (const float*)d_in[20], (const float*)d_in[21], (const float*)d_in[22], (const float*)d_in[23],
        (const float*)d_in[24], (const float*)d_in[25], (const float*)d_in[26], (const float*)d_in[27],
        (const float*)d_in[28], (const float*)d_in[29], (const float*)d_in[30], (const float*)d_in[31],
        (const float*)d_in[32], (const float*)d_in[33], (const float*)d_in[34], (const float*)d_in[35],
        out);
    (void)in_sizes; (void)n_in; (void)out_size; (void)ws_size;
}
